// Attention_77094662963277
// MI455X (gfx1250) — matrix operand order repaired and checked by reading, not compiled, not run on hardware
//
#include <hip/hip_runtime.h>
#include <hip/hip_bf16.h>

typedef __attribute__((ext_vector_type(16))) __bf16 v16bf;
typedef __attribute__((ext_vector_type(4)))  __bf16 bf16x4;
typedef __attribute__((ext_vector_type(2)))  __bf16 bf16x2;
typedef __attribute__((ext_vector_type(8)))  float  v8f;
typedef __attribute__((ext_vector_type(4)))  float  f32x4;
typedef __attribute__((ext_vector_type(4)))  unsigned int v4u;
typedef __attribute__((ext_vector_type(8)))  unsigned int v8u;

#define GS 40   // bf16 per channel-row in kernel1 staging (32 l + 8 pad); 80 B rows
#define XS 72   // bf16 per pixel-row   in kernel3 staging (64 ch + 8 pad); 144 B rows
#define AS 72   // bf16 per out-ch row  in kernel3 A staging (64 k + 8 pad)

// Assemble a 16x32 bf16 WMMA fragment from two 16-byte LDS loads.
__device__ __forceinline__ v16bf load_frag2(const unsigned int* p0,
                                            const unsigned int* p1) {
  v8u u;
  u.lo = *(const v4u*)p0;
  u.hi = *(const v4u*)p1;
  return __builtin_bit_cast(v16bf, u);
}

// ---------------------------------------------------------------------------
// Kernel 1: per-block Gram matrices G = X^T X, X = [16384, 64] per (b, n).
// Grid: 256 WGs = 32 blocks x 8 slices; 8 waves/WG, each wave covers 256 rows.
// LDS staging is channel-major [c][l] (fragment K-pairs packed as dwords,
// fragments = 2x ds_load_b128).  l-quads are XOR-swizzled by (c>>4)&3 so the
// transpose stores are bank-conflict-free.
// ---------------------------------------------------------------------------
__global__ __launch_bounds__(256) void ca_gram(const float* __restrict__ x,
                                               float* __restrict__ gpart) {
  __shared__ __bf16 sm[8 * 64 * GS];   // per-wave: 64 ch x 40 (5120 B) -> 40 KB
  __shared__ float  gred[4096];        // flat-layout reduction buffer (16 KB)

  const int tid  = threadIdx.x;
  const int w    = tid >> 5, lane = tid & 31;
  const int half = lane >> 4, m   = lane & 15;
  const int wg    = blockIdx.x;
  const int slice = wg & 7, blk = wg >> 3;
  const int bb = blk >> 2, nn = blk & 3;
  const int m1 = nn >> 1, m2 = nn & 1;

  v8f acc[16];
#pragma unroll
  for (int i = 0; i < 16; ++i) acc[i] = (v8f){0.f,0.f,0.f,0.f,0.f,0.f,0.f,0.f};

  __bf16* smw = sm + w * 64 * GS;
  unsigned int* smu = (unsigned int*)smw;        // 64 rows x 20 uints
  const size_t imgBase = (size_t)bb * 256 * 256 * 64;

  const int h = half;          // row-select within staging pass
  const int q = m;             // channel quad 0..15

  for (int it = 0; it < 8; ++it) {
    const int l0 = slice * 2048 + w * 256 + it * 32;
    // stage 32 l-rows x 64 ch -> [c][l] bf16, l-pairs packed, quads swizzled
#pragma unroll
    for (int s = 0; s < 8; ++s) {
      int l  = l0 + 4 * s + 2 * h;
      int gy = ((l >> 7) << 1) + m1;          // y = 2*hy + m1
      int gx = ((l & 127) << 1) + m2;         // x = 2*wx + m2
      const float* r0 = x + imgBase + ((size_t)gy * 256 + gx) * 64 + 4 * q;
      float4 a0 = *(const float4*)r0;         // row l   , ch 4q..4q+3
      float4 a1 = *(const float4*)(r0 + 128); // row l+1 (+2 px)
      const float* e0 = (const float*)&a0;
      const float* e1 = (const float*)&a1;
      int lpos = 2 * s + h;                   // l-pair slot 0..15
      int quad = lpos >> 2, rem = lpos & 3;
      int sw   = (q >> 2) & 3;                // = (c>>4)&3
#pragma unroll
      for (int j = 0; j < 4; ++j) {
        int c = 4 * q + j;
        bf16x2 pr = { (__bf16)e0[j], (__bf16)e1[j] };
        *(bf16x2*)(smu + c * 20 + ((quad ^ sw) << 2) + rem) = pr;
      }
    }
    __syncthreads();

    // fragments: A[c_m, l_k] = X[l,c], B[l_k, d_n] = X[l,d]; both from [c][l]
    v16bf af[4], bfv[4];
#pragma unroll
    for (int t4 = 0; t4 < 4; ++t4) {
      const unsigned int* rb = smu + (t4 * 16 + m) * 20;
      af[t4]  = load_frag2(rb + (((half    ) ^ t4) << 2),
                           rb + (((half + 2) ^ t4) << 2));
      bfv[t4] = load_frag2(rb + (((half    ) ^ t4) << 2),
                           rb + (((half + 2) ^ t4) << 2));
    }
#pragma unroll
    for (int ct = 0; ct < 4; ++ct)
#pragma unroll
      for (int dt = 0; dt < 4; ++dt)
        acc[ct*4+dt] = __builtin_amdgcn_wmma_f32_16x16x32_bf16(
            false, af[ct], false, bfv[dt], (short)0, acc[ct*4+dt], false, false);
    __syncthreads();
  }

  // deterministic cross-wave reduction, flat per-lane layout, b128 LDS ops.
  // Waves 0..6 accumulate into LDS in fixed order; wave 7 adds the running
  // sum into its registers and writes the partial Gram straight to global.
  if (w == 0) {
#pragma unroll
    for (int t = 0; t < 16; ++t)
#pragma unroll
      for (int g = 0; g < 2; ++g) {
        f32x4 val = { acc[t][g*4+0], acc[t][g*4+1], acc[t][g*4+2], acc[t][g*4+3] };
        *(f32x4*)(gred + (t * 2 + g) * 128 + lane * 4) = val;
      }
  }
  __syncthreads();
  for (int ww = 1; ww < 7; ++ww) {
    if (w == ww) {
#pragma unroll
      for (int t = 0; t < 16; ++t)
#pragma unroll
        for (int g = 0; g < 2; ++g) {
          f32x4* p = (f32x4*)(gred + (t * 2 + g) * 128 + lane * 4);
          f32x4 cur = *p;
          f32x4 val = { acc[t][g*4+0], acc[t][g*4+1], acc[t][g*4+2], acc[t][g*4+3] };
          *p = cur + val;
        }
    }
    __syncthreads();
  }
  if (w == 7) {
    float* gp = gpart + (size_t)wg * 4096;
#pragma unroll
    for (int ct = 0; ct < 4; ++ct)
#pragma unroll
      for (int dt = 0; dt < 4; ++dt) {
        int t = ct * 4 + dt;
#pragma unroll
        for (int g = 0; g < 2; ++g) {
          f32x4 cur = *(const f32x4*)(gred + (t * 2 + g) * 128 + lane * 4);
#pragma unroll
          for (int j = 0; j < 4; ++j) {
            int vg = g * 4 + j;
            int c  = ct * 16 + vg + half * 8;   // D: M = vgpr + 8*(lane>=16)
            int d  = dt * 16 + m;               //    N = lane%16
            gp[c * 64 + d] = cur[j] + acc[t][vg];
          }
        }
      }
  }
}

// ---------------------------------------------------------------------------
// Kernel 2: per block (32 WGs):  G -> score -> softmax -> A = Wv W^T Wproj
// f32 VALU math with float4 vector loads/FMAs; two aliased LDS buffers.
// ---------------------------------------------------------------------------
__global__ __launch_bounds__(256) void ca_combine(const float* __restrict__ gpart,
                                                  const float* __restrict__ Wqkv,
                                                  const float* __restrict__ Wproj,
                                                  float* __restrict__ Aout) {
  __shared__ float B0[4096];  // G, then S/weight
  __shared__ float B1[4096];  // T, then P
  const int tid = threadIdx.x;
  const int blk = blockIdx.x;
  const int row = tid >> 2;
  const int cb  = (tid & 3) * 16;
  f32x4 a4[4];

  // 1) G = sum of 8 partials (b128 traffic)
#pragma unroll
  for (int v = 0; v < 4; ++v) {
    int idx = tid * 16 + v * 4;
    f32x4 s = {0.f,0.f,0.f,0.f};
#pragma unroll
    for (int p = 0; p < 8; ++p)
      s += *(const f32x4*)(gpart + (size_t)(blk * 8 + p) * 4096 + idx);
    *(f32x4*)(B0 + idx) = s;
  }
  __syncthreads();

  // 2) T = G @ Wk            (Wk = Wqkv[:, 64:128])
#pragma unroll
  for (int i = 0; i < 4; ++i) a4[i] = (f32x4){0.f,0.f,0.f,0.f};
  for (int f = 0; f < 64; ++f) {
    float g = B0[row * 64 + f];
    f32x4 gv = {g, g, g, g};
    const f32x4* wr = (const f32x4*)(Wqkv + f * 192 + 64 + cb);
#pragma unroll
    for (int i = 0; i < 4; ++i) a4[i] += gv * wr[i];
  }
#pragma unroll
  for (int i = 0; i < 4; ++i) *(f32x4*)(B1 + row * 64 + cb + i * 4) = a4[i];
  __syncthreads();

  // 3) S = Wq^T @ T * c^-0.5 (Wq = Wqkv[:, 0:64]); overwrite B0
#pragma unroll
  for (int i = 0; i < 4; ++i) a4[i] = (f32x4){0.f,0.f,0.f,0.f};
  for (int e = 0; e < 64; ++e) {
    float wq = Wqkv[e * 192 + row];
    f32x4 wv = {wq, wq, wq, wq};
    const f32x4* tr = (const f32x4*)(B1 + e * 64 + cb);
#pragma unroll
    for (int i = 0; i < 4; ++i) a4[i] += wv * tr[i];
  }
#pragma unroll
  for (int i = 0; i < 4; ++i) {
    f32x4 sc = {0.125f, 0.125f, 0.125f, 0.125f};
    *(f32x4*)(B0 + row * 64 + cb + i * 4) = a4[i] * sc;
  }
  __syncthreads();

  // 4) row softmax on B0
  if (tid < 64) {
    float mx = -3.4e38f;
    for (int d = 0; d < 64; ++d) mx = fmaxf(mx, B0[tid * 64 + d]);
    float s = 0.f;
    for (int d = 0; d < 64; ++d) {
      float e2 = __expf(B0[tid * 64 + d] - mx);
      B0[tid * 64 + d] = e2; s += e2;
    }
    float inv = 1.f / s;
    for (int d = 0; d < 64; ++d) B0[tid * 64 + d] *= inv;
  }
  __syncthreads();

  // 5) P = weight^T @ Wproj; overwrite B1.   row acts as d
#pragma unroll
  for (int i = 0; i < 4; ++i) a4[i] = (f32x4){0.f,0.f,0.f,0.f};
  for (int c2 = 0; c2 < 64; ++c2) {
    float wg = B0[c2 * 64 + row];
    f32x4 wv = {wg, wg, wg, wg};
    const f32x4* pr = (const f32x4*)(Wproj + c2 * 64 + cb);
#pragma unroll
    for (int i = 0; i < 4; ++i) a4[i] += wv * pr[i];
  }
#pragma unroll
  for (int i = 0; i < 4; ++i) *(f32x4*)(B1 + row * 64 + cb + i * 4) = a4[i];
  __syncthreads();

  // 6) A = Wv @ P            (Wv = Wqkv[:, 128:192]); row acts as e
#pragma unroll
  for (int i = 0; i < 4; ++i) a4[i] = (f32x4){0.f,0.f,0.f,0.f};
  for (int d = 0; d < 64; ++d) {
    float wv = Wqkv[row * 192 + 128 + d];
    f32x4 wvv = {wv, wv, wv, wv};
    const f32x4* pr = (const f32x4*)(B1 + d * 64 + cb);
#pragma unroll
    for (int i = 0; i < 4; ++i) a4[i] += wvv * pr[i];
  }
#pragma unroll
  for (int i = 0; i < 4; ++i)
    *(f32x4*)(Aout + (size_t)blk * 4096 + row * 64 + cb + i * 4) = a4[i];
}

// ---------------------------------------------------------------------------
// Kernel 3: y[l,:] = x[l,:] @ A_{b,n}.  One WG per (b, image row y).
// x staged row-major (K = channel minor), A staged transposed [out][in]
// (K = in-channel minor) so all fragments load as 2x ds_load_b128.
// ---------------------------------------------------------------------------
__global__ __launch_bounds__(256) void ca_apply(const float* __restrict__ x,
                                                const float* __restrict__ Aall,
                                                float* __restrict__ y) {
  __shared__ __bf16 xrow[256 * XS];      // 36,864 B
  __shared__ __bf16 Asm[2 * 64 * AS];    // 18,432 B

  const int tid  = threadIdx.x;
  const int w    = tid >> 5, lane = tid & 31;
  const int half = lane >> 4, m   = lane & 15;
  const int bb = blockIdx.x >> 8;
  const int yy = blockIdx.x & 255;
  const int m1 = yy & 1;

  // stage both A matrices (m2 = 0,1), transposed to [n=out][k=in], bf16 pairs
  for (int n2 = 0; n2 < 2; ++n2) {
    const float* Ab = Aall + (size_t)(bb * 4 + m1 * 2 + n2) * 4096;
    __bf16* As = Asm + n2 * 64 * AS;
    for (int j = 0; j < 8; ++j) {
      int idx = j * 256 + tid;        // 0..2047 = (e-pair, f)
      int f  = idx & 63;
      int ep = idx >> 6;
      bf16x2 p = { (__bf16)Ab[(2 * ep) * 64 + f],
                   (__bf16)Ab[(2 * ep + 1) * 64 + f] };
      *(bf16x2*)(As + f * AS + 2 * ep) = p;
    }
  }
  // stage image row (256 px x 64 ch) as bf16, packed b64 stores
  const float* xr = x + ((size_t)(bb * 256 + yy) * 256) * 64;
  for (int p = 0; p < 16; ++p) {
    int gi = (p * 256 + tid) * 4;
    int px = gi >> 6, ch = gi & 63;
    float4 v4 = *(const float4*)(xr + gi);
    bf16x4 qv = { (__bf16)v4.x, (__bf16)v4.y, (__bf16)v4.z, (__bf16)v4.w };
    *(bf16x4*)(xrow + px * XS + ch) = qv;
  }
  __syncthreads();

  const unsigned int* xu = (const unsigned int*)xrow;   // px rows x 36 uints
  float* yb = y + ((size_t)(bb * 256 + yy) * 256) * 64;

  for (int it = 0; it < 2; ++it) {
    int task   = it * 8 + w;       // 16 tasks: parity x 8 pixel groups
    int parity = task & 1;
    int grp    = task >> 1;

    int px = ((grp * 16 + m) << 1) + parity;
    const unsigned int* ap = xu + px * (XS / 2) + half * 4;
    v16bf axf[2];
#pragma unroll
    for (int kh = 0; kh < 2; ++kh)
      axf[kh] = load_frag2(ap + kh * 16, ap + kh * 16 + 8);

    const unsigned int* au = (const unsigned int*)(Asm + parity * 64 * AS);
#pragma unroll
    for (int nt = 0; nt < 4; ++nt) {
      v8f acc = (v8f){0.f,0.f,0.f,0.f,0.f,0.f,0.f,0.f};
#pragma unroll
      for (int kh = 0; kh < 2; ++kh) {
        const unsigned int* bu =
            au + (nt * 16 + m) * (AS / 2) + kh * 16 + half * 4;
        v16bf bfr = load_frag2(bu, bu + 8);
        acc = __builtin_amdgcn_wmma_f32_16x16x32_bf16(
            false, axf[kh], false, bfr, (short)0, acc, false, false);
      }
#pragma unroll
      for (int vg = 0; vg < 8; ++vg) {
        int mm  = vg + half * 8;
        int pxs = ((grp * 16 + mm) << 1) + parity;
        yb[(size_t)pxs * 64 + nt * 16 + m] = acc[vg];
      }
    }
  }
}

// ---------------------------------------------------------------------------
extern "C" void kernel_launch(void* const* d_in, const int* in_sizes, int n_in,
                              void* d_out, int out_size, void* d_ws, size_t ws_size,
                              hipStream_t stream) {
  const float* x     = (const float*)d_in[0];   // [8,256,256,64]
  const float* Wqkv  = (const float*)d_in[1];   // [64,192]
  const float* Wproj = (const float*)d_in[2];   // [64,64]
  float* out = (float*)d_out;

  float* gpart = (float*)d_ws;                  // 256 * 4096 f32 partial Grams
  float* Amats = gpart + 256 * 4096;            // 32  * 4096 f32 A matrices

  ca_gram   <<<256,  256, 0, stream>>>(x, gpart);
  ca_combine<<<32,   256, 0, stream>>>(gpart, Wqkv, Wproj, Amats);
  ca_apply  <<<2048, 256, 0, stream>>>(x, Amats, out);
}
